// DiscreteEntropyComputer_88304527606072
// MI455X (gfx1250) — compile-verified, matrix operand order checked by reading, not run on hardware
//
#include <hip/hip_runtime.h>

// ---------------------------------------------------------------------------
// DiscreteEntropyComputer for MI455X (gfx1250, wave32)
//
// values: [256, 262144] f32  ->  entropy: [256] f32
// Memory-bound: 268 MB read @ 23.3 TB/s => ~11.5 us floor.
// Path: b128 NT global loads -> sign->u8 pack -> V_WMMA_I32_16X16X64_IU8 with
// all-ones B as a 1024-way adder tree -> LDS block reduce -> tiny finalize.
// ---------------------------------------------------------------------------

typedef __attribute__((ext_vector_type(8))) int   v8i;
typedef __attribute__((ext_vector_type(4))) float v4f;

#define ROWS     256
#define ROW_LEN  262144              // elements per row
#define SPLIT    8                   // blocks per row
#define SEG      (ROW_LEN / SPLIT)   // 32768 elements per block
#define BLOCK    256                 // 8 wave32s per block
#define LANE_ELEMS 32                // elements packed per lane per WMMA
#define WAVE_ELEMS (32 * LANE_ELEMS)           // 1024 per wave per WMMA
#define BLOCK_ITER (BLOCK * LANE_ELEMS)        // 8192 per block iteration
#define N_ITERS  (SEG / BLOCK_ITER)            // 4

__global__ void __launch_bounds__(BLOCK)
count_pos_wmma(const float* __restrict__ x, int* __restrict__ partials) {
    const int row  = blockIdx.x / SPLIT;
    const int part = blockIdx.x % SPLIT;
    const float* rowp = x + (size_t)row * ROW_LEN + (size_t)part * SEG;

    const int lane = threadIdx.x & 31;
    const int wave = threadIdx.x >> 5;

    // B = all-ones u8 matrix: every column of D accumulates the full row-sum
    // of A, so the reduction is independent of the A register layout.
    v8i ones;
#pragma unroll
    for (int i = 0; i < 8; ++i) ones[i] = 0x01010101;

    v8i acc;
#pragma unroll
    for (int i = 0; i < 8; ++i) acc[i] = 0;

#pragma unroll
    for (int it = 0; it < N_ITERS; ++it) {
        const float* base = rowp + it * BLOCK_ITER + wave * WAVE_ELEMS;
        v8i a;
#pragma unroll
        for (int r = 0; r < 8; ++r) {
            // Coalesced: 32 lanes x 16B = 512B contiguous per round.
            // Non-temporal: stream (256MB) > L2 (192MB), read exactly once.
            const v4f v = __builtin_nontemporal_load(
                (const v4f*)(base + (size_t)(r * 32 + lane) * 4));
            unsigned b0 = v.x > 0.0f;
            unsigned b1 = v.y > 0.0f;
            unsigned b2 = v.z > 0.0f;
            unsigned b3 = v.w > 0.0f;
            a[r] = (int)(b0 | (b1 << 8) | (b2 << 16) | (b3 << 24));
        }
        // D = A(16x64 u8) * ones(64x16 u8) + C : folds 1024 quantized
        // elements into the i32 accumulator tile per instruction.
        acc = __builtin_amdgcn_wmma_i32_16x16x64_iu8(
            /*sgn_a=*/false, a, /*sgn_b=*/false, ones, acc,
            /*reuse_a=*/false, /*reuse_b=*/false);
    }

    // Sum this lane's 8 accumulator VGPRs; summing over all 32 lanes gives
    // Sum(all 256 D entries) = 16 * (elements counted by this wave).
    int p = 0;
#pragma unroll
    for (int i = 0; i < 8; ++i) p += acc[i];

    __shared__ int red[BLOCK];
    red[threadIdx.x] = p;
    __syncthreads();
#pragma unroll
    for (int s = BLOCK / 2; s > 0; s >>= 1) {
        if (threadIdx.x < s) red[threadIdx.x] += red[threadIdx.x + s];
        __syncthreads();
    }
    if (threadIdx.x == 0) {
        // Every element was counted 16x (16 identical D columns).
        partials[row * SPLIT + part] = red[0] >> 4;
    }
}

__global__ void finalize_entropy(const int* __restrict__ partials,
                                 float* __restrict__ out) {
    const int row = blockIdx.x * blockDim.x + threadIdx.x;
    if (row >= ROWS) return;

    int c1 = 0;
#pragma unroll
    for (int i = 0; i < SPLIT; ++i) c1 += partials[row * SPLIT + i];
    const int c0 = ROW_LEN - c1;

    const float denom = (float)ROW_LEN + 1e-8f;   // matches reference epsilon
    const float p0 = (float)c0 / denom;
    const float p1 = (float)c1 / denom;
    const float t0 = (p0 > 0.0f) ? p0 * log2f(p0 + 1e-10f) : 0.0f;
    const float t1 = (p1 > 0.0f) ? p1 * log2f(p1 + 1e-10f) : 0.0f;
    out[row] = -(t0 + t1);
}

extern "C" void kernel_launch(void* const* d_in, const int* in_sizes, int n_in,
                              void* d_out, int out_size, void* d_ws, size_t ws_size,
                              hipStream_t stream) {
    const float* values = (const float*)d_in[0];
    float*       out    = (float*)d_out;
    int*         parts  = (int*)d_ws;   // ROWS * SPLIT * 4B = 8 KB, written
                                        // fully before read every call.

    count_pos_wmma<<<ROWS * SPLIT, BLOCK, 0, stream>>>(values, parts);
    finalize_entropy<<<1, BLOCK, 0, stream>>>(parts, out);
}
